// BiMambaBlock_61529701482996
// MI455X (gfx1250) — compile-verified, matrix operand order checked by reading, not run on hardware
//
#include <hip/hip_runtime.h>
#include <hip/hip_bf16.h>

// ---------------------------------------------------------------------------
// BiMambaBlock for MI455X (gfx1250, wave32, WMMA + TDM).
//   P0  cvt       : x and all weights -> bf16 arena (one-time)
//   K1  gemm FULL : xz = x @ in_proj_w^T (fp32 out)          [TDM-staged]
//   K2  conv_silu : u = silu(depthwise causal conv(u)) -> bf16
//   K3  gemm SAFE : x_dbl = u @ x_proj_w^T (N=48) -> fp32 + bf16
//   K4  gemm SAFE : dt = softplus(x_dbl[:,:16] @ dt_w^T + b) (K=16) -> fp32
//   K5  scan      : selective scan (dA/dBu recomputed in-reg), gate with
//                   silu(z), pipelined loads, y -> bf16
//   K6  gemm FULL : h = y_f@fW^T + y_b@bW^T + x (dual pass)  [TDM-staged]
//   K7  layernorm : out = LN(h)*gamma + beta
// Workspace: ~224 MB fp32 + ~81 MB bf16.
// ---------------------------------------------------------------------------

#define D_MODEL  256
#define D_STATE  16
#define D_CONV   4
#define D_INNER  512
#define DT_RANK  16
#define BATCH    8
#define SEQ      2048
#define NTOK     (BATCH * SEQ)           // 16384
#define XDBL_W   (DT_RANK + 2 * D_STATE) // 48
#define LN_EPS   1e-5f

#define BM 128
#define BN 64
#define KT 32
#define LDS_STRIDE 40   // bf16 elems/row = 80 B: 16B-aligned rows, no bank conflicts

typedef __attribute__((ext_vector_type(16))) __bf16 v16bf;
typedef __attribute__((ext_vector_type(8)))  float  v8f;
typedef __attribute__((ext_vector_type(4)))  unsigned uint32x4;
typedef __attribute__((ext_vector_type(8)))  int      int32x8;
typedef __attribute__((ext_vector_type(4)))  int      int32x4;

#if defined(__has_builtin)
#  if __has_builtin(__builtin_amdgcn_tensor_load_to_lds) && \
      __has_builtin(__builtin_amdgcn_s_wait_tensorcnt)
#    define ENABLE_TDM 1
#  endif
#endif
#ifndef ENABLE_TDM
#  define ENABLE_TDM 0
#endif

static __device__ __forceinline__ unsigned short bf16bits(float f) {
    unsigned u = __builtin_bit_cast(unsigned, f);
    u += 0x7FFFu + ((u >> 16) & 1u);                // RNE
    return (unsigned short)(u >> 16);
}
static __device__ __forceinline__ __bf16 f2bf(float f) {
    unsigned short s = bf16bits(f);
    return __builtin_bit_cast(__bf16, s);
}
static __device__ __forceinline__ float bf2f(__bf16 h) {
    unsigned u = ((unsigned)__builtin_bit_cast(unsigned short, h)) << 16;
    return __builtin_bit_cast(float, u);
}
static __device__ __forceinline__ __bf16 bf16_zero() {
    return __builtin_bit_cast(__bf16, (unsigned short)0);
}
static __device__ __forceinline__ float sigmoid_(float v) {
    return 1.0f / (1.0f + __expf(-v));
}
static __device__ __forceinline__ float softplus_(float v) {
    return (v > 20.0f) ? v : __logf(1.0f + __expf(v));
}

#if ENABLE_TDM
// Issue one TDM 2D tile load: rows x 32 bf16 tile, row stride ld_elems, into
// LDS at lds_byte with +16B padding every 64B so LDS row stride = 80 B.
static __device__ __forceinline__ void tdm_load_tile(unsigned lds_byte,
        const __bf16* gsrc, int ld_elems, int rows)
{
    unsigned long long ga = (unsigned long long)(size_t)gsrc;
    uint32x4 g0;
    g0[0] = 1u;                                            // count=1, user D#
    g0[1] = lds_byte;                                      // lds_addr
    g0[2] = (unsigned)ga;                                  // global_addr[31:0]
    g0[3] = (unsigned)((ga >> 32) & 0x01FFFFFFu) | 0x80000000u; // ga[56:32]|type=2
    int32x8 g1;
    g1[0] = (1 << 16) | (1 << 20) | (3 << 22) | (3 << 25); // 2B elems, pad: 16DW int, 4DW amt
    g1[1] = (KT & 0xFFFF) << 16;                           // tensor_dim0 = 32
    g1[2] = (rows & 0xFFFF) << 16;                         // tensor_dim1 lo = rows
    g1[3] = (KT & 0xFFFF) << 16;                           // tile_dim0 = 32
    g1[4] = rows & 0xFFFF;                                 // tile_dim1 = rows
    g1[5] = ld_elems;                                      // tensor_dim0_stride lo
    g1[6] = 0;
    g1[7] = 0;
    int32x4 z4; z4[0] = z4[1] = z4[2] = z4[3] = 0;
#if __clang_major__ >= 23
    int32x8 z8; for (int i = 0; i < 8; ++i) z8[i] = 0;
    __builtin_amdgcn_tensor_load_to_lds(g0, g1, z4, z4, z8, 0);
#else
    __builtin_amdgcn_tensor_load_to_lds(g0, g1, z4, z4, 0);
#endif
}
#endif

// Vectorized fallback staging: ROWS x 32 bf16 tile via b128 copies.
template<int ROWS>
static __device__ __forceinline__ void stage_tile(const __bf16* __restrict__ src,
                                                  int ld, __bf16* __restrict__ dst, int tid)
{
#pragma unroll
    for (int it = 0; it < (ROWS * 4) / 256; ++it) {
        const int i = tid + it * 256;
        const int r = i >> 2, c = i & 3;
        uint4 v = *reinterpret_cast<const uint4*>(src + (size_t)r * ld + c * 8);
        *reinterpret_cast<uint4*>(dst + r * LDS_STRIDE + c * 8) = v;
    }
}

// One K-stage of 2x2 wmma tiles from staged LDS buffers.
static __device__ __forceinline__ void mma_stage(const __bf16* __restrict__ As,
        const __bf16* __restrict__ Bs, int wm, int wn, int lrow, int kbase,
        v8f acc[2][2])
{
    v16bf afrag[2], bfrag[2];
#pragma unroll
    for (int mt = 0; mt < 2; ++mt) {
        const int r = wm + mt * 16 + lrow;
#pragma unroll
        for (int p = 0; p < 8; ++p) {
            const int kk = (p < 4) ? (kbase + 2 * p) : (16 + kbase + 2 * (p - 4));
            union { unsigned u; __bf16 h[2]; } pk;
            pk.u = *reinterpret_cast<const unsigned*>(&As[r * LDS_STRIDE + kk]);
            afrag[mt][2 * p]     = pk.h[0];
            afrag[mt][2 * p + 1] = pk.h[1];
        }
    }
#pragma unroll
    for (int nt = 0; nt < 2; ++nt) {
        const int r = wn + nt * 16 + lrow;
#pragma unroll
        for (int p = 0; p < 8; ++p) {
            const int kk = (p < 4) ? (kbase + 2 * p) : (16 + kbase + 2 * (p - 4));
            union { unsigned u; __bf16 h[2]; } pk;
            pk.u = *reinterpret_cast<const unsigned*>(&Bs[r * LDS_STRIDE + kk]);
            bfrag[nt][2 * p]     = pk.h[0];
            bfrag[nt][2 * p + 1] = pk.h[1];
        }
    }
#pragma unroll
    for (int mt = 0; mt < 2; ++mt)
#pragma unroll
        for (int nt = 0; nt < 2; ++nt)
            acc[mt][nt] = __builtin_amdgcn_wmma_f32_16x16x32_bf16(
                false, afrag[mt], false, bfrag[nt], (short)0, acc[mt][nt], false, false);
}

// ---------------------------------------------------------------------------
// FULL GEMM: bf16 operands, K%32==0, N%64==0, lda dense. Optional second
// A2@W2^T pass summed in, optional fp32 residual. TDM double-buffered staging.
// ---------------------------------------------------------------------------
__global__ __launch_bounds__(256, 2)
void gemm_bf16_full_kernel(const __bf16* __restrict__ A,  const __bf16* __restrict__ W,
                           const __bf16* __restrict__ A2, const __bf16* __restrict__ W2,
                           const float* __restrict__ resid, float* __restrict__ C,
                           int M, int N, int K, int lda)
{
    __shared__ __attribute__((aligned(16))) __bf16 As[2][BM * LDS_STRIDE];
    __shared__ __attribute__((aligned(16))) __bf16 Bs[2][BN * LDS_STRIDE];

    const int tid   = threadIdx.x;
    const int lane  = tid & 31;
    const int wave  = tid >> 5;
    const int wm    = (wave & 3) * 32;
    const int wn    = (wave >> 2) * 32;
    const int m0    = blockIdx.y * BM;
    const int n0    = blockIdx.x * BN;
    const int lrow  = lane & 15;
    const int lhalf = lane >> 4;
    const int kbase = lhalf * 8;

    v8f acc[2][2];
#pragma unroll
    for (int i = 0; i < 2; ++i)
#pragma unroll
        for (int j = 0; j < 2; ++j)
#pragma unroll
            for (int r = 0; r < 8; ++r) acc[i][j][r] = 0.0f;

    const int nk = K / KT;
    const int nstage = ((A2 != nullptr) ? 2 : 1) * nk;

#if ENABLE_TDM
    auto tdm_issue = [&](int s) {
        const int pass = s / nk, kt = s - pass * nk;
        const int buf = s & 1;
        const __bf16* Ap = pass ? A2 : A;
        const __bf16* Wp = pass ? W2 : W;
        tdm_load_tile((unsigned)(size_t)&As[buf][0],
                      Ap + (size_t)m0 * lda + (size_t)kt * KT, lda, BM);
        tdm_load_tile((unsigned)(size_t)&Bs[buf][0],
                      Wp + (size_t)n0 * K + (size_t)kt * KT, K, BN);
    };
    if (wave == 0) tdm_issue(0);
    for (int s = 0; s < nstage; ++s) {
        const int buf = s & 1;
        if (wave == 0) {
            if (s + 1 < nstage) {
                tdm_issue(s + 1);                       // prefetch next stage
                __builtin_amdgcn_s_wait_tensorcnt((short)2);  // current done
            } else {
                __builtin_amdgcn_s_wait_tensorcnt((short)0);
            }
        }
        __syncthreads();
        mma_stage(&As[buf][0], &Bs[buf][0], wm, wn, lrow, kbase, acc);
        __syncthreads();
    }
#else
    for (int s = 0; s < nstage; ++s) {
        const int pass = s / nk, kt = s - pass * nk;
        const __bf16* Ap = pass ? A2 : A;
        const __bf16* Wp = pass ? W2 : W;
        stage_tile<BM>(Ap + (size_t)m0 * lda + (size_t)kt * KT, lda, &As[0][0], tid);
        stage_tile<BN>(Wp + (size_t)n0 * K + (size_t)kt * KT, K, &Bs[0][0], tid);
        __syncthreads();
        mma_stage(&As[0][0], &Bs[0][0], wm, wn, lrow, kbase, acc);
        __syncthreads();
    }
#endif

#pragma unroll
    for (int mt = 0; mt < 2; ++mt)
#pragma unroll
        for (int nt = 0; nt < 2; ++nt) {
            const int gn = n0 + wn + nt * 16 + lrow;
#pragma unroll
            for (int r = 0; r < 8; ++r) {
                const int gm = m0 + wm + mt * 16 + 8 * lhalf + r;
                float v = acc[mt][nt][r];
                if (resid) v += resid[(size_t)gm * N + gn];
                C[(size_t)gm * N + gn] = v;
            }
        }
}

// ---------------------------------------------------------------------------
// SAFE GEMM: bf16 operands, guarded for partial N (x_proj N=48) and partial
// K (dt_proj K=16, lda=48). Optional bias / softplus / bf16 shadow output.
// ---------------------------------------------------------------------------
__global__ __launch_bounds__(256, 2)
void gemm_bf16_safe_kernel(const __bf16* __restrict__ A, const __bf16* __restrict__ W,
                           const float* __restrict__ bias, float* __restrict__ C,
                           __bf16* __restrict__ Cbf,
                           int M, int N, int K, int lda, int act)
{
    __shared__ __attribute__((aligned(16))) __bf16 As[BM * LDS_STRIDE];
    __shared__ __attribute__((aligned(16))) __bf16 Bs[BN * LDS_STRIDE];

    const int tid   = threadIdx.x;
    const int lane  = tid & 31;
    const int wave  = tid >> 5;
    const int wm    = (wave & 3) * 32;
    const int wn    = (wave >> 2) * 32;
    const int m0    = blockIdx.y * BM;
    const int n0    = blockIdx.x * BN;
    const int lrow  = lane & 15;
    const int lhalf = lane >> 4;
    const int kbase = lhalf * 8;

    v8f acc[2][2];
#pragma unroll
    for (int i = 0; i < 2; ++i)
#pragma unroll
        for (int j = 0; j < 2; ++j)
#pragma unroll
            for (int r = 0; r < 8; ++r) acc[i][j][r] = 0.0f;

    for (int k0 = 0; k0 < K; k0 += KT) {
#pragma unroll
        for (int it = 0; it < (BM * KT) / 256; ++it) {
            const int i  = tid + it * 256;
            const int r  = i >> 5, kk = i & 31;
            const int gk = k0 + kk;
            As[r * LDS_STRIDE + kk] =
                (gk < K) ? A[(size_t)(m0 + r) * lda + gk] : bf16_zero();
        }
#pragma unroll
        for (int it = 0; it < (BN * KT) / 256; ++it) {
            const int i  = tid + it * 256;
            const int r  = i >> 5, kk = i & 31;
            const int gk = k0 + kk, gn = n0 + r;
            Bs[r * LDS_STRIDE + kk] =
                (gn < N && gk < K) ? W[(size_t)gn * K + gk] : bf16_zero();
        }
        __syncthreads();
        mma_stage(As, Bs, wm, wn, lrow, kbase, acc);
        __syncthreads();
    }

#pragma unroll
    for (int mt = 0; mt < 2; ++mt)
#pragma unroll
        for (int nt = 0; nt < 2; ++nt) {
            const int gn = n0 + wn + nt * 16 + lrow;
            if (gn >= N) continue;
#pragma unroll
            for (int r = 0; r < 8; ++r) {
                const int gm = m0 + wm + mt * 16 + 8 * lhalf + r;
                float v = acc[mt][nt][r];
                if (bias) v += bias[gn];
                if (act == 1) v = softplus_(v);
                C[(size_t)gm * N + gn] = v;
                if (Cbf) Cbf[(size_t)gm * N + gn] = f2bf(v);
            }
        }
}

// ---------------------------------------------------------------------------
// fp32 -> bf16 bulk convert (float4 -> 4x bf16 packed stores).
// ---------------------------------------------------------------------------
__global__ __launch_bounds__(256)
void cvt_f32_bf16_kernel(const float* __restrict__ in, __bf16* __restrict__ out, int n4)
{
    const int i = blockIdx.x * 256 + threadIdx.x;
    if (i >= n4) return;
    float4 v = reinterpret_cast<const float4*>(in)[i];
    uint2 o;
    o.x = (unsigned)bf16bits(v.x) | ((unsigned)bf16bits(v.y) << 16);
    o.y = (unsigned)bf16bits(v.z) | ((unsigned)bf16bits(v.w) << 16);
    reinterpret_cast<uint2*>(out)[i] = o;
}

// ---------------------------------------------------------------------------
// Causal depthwise conv1d (4 taps) + bias + SiLU -> bf16 u buffer.
// ---------------------------------------------------------------------------
__global__ __launch_bounds__(256)
void conv_silu_kernel(const float* __restrict__ xz, const float* __restrict__ cw,
                      const float* __restrict__ cb, __bf16* __restrict__ ucbf, int rev)
{
    const int idx = blockIdx.x * blockDim.x + threadIdx.x;
    if (idx >= NTOK * D_INNER) return;
    const int d = idx % D_INNER;
    const int s = (idx / D_INNER) % SEQ;
    const int b = idx / (D_INNER * SEQ);

    float acc = cb[d];
#pragma unroll
    for (int j = 0; j < D_CONV; ++j) {
        const int off = D_CONV - 1 - j;
        const int t   = rev ? (s + off) : (s - off);
        if (t >= 0 && t < SEQ)
            acc += cw[d * D_CONV + j] *
                   xz[((size_t)(b * SEQ + t)) * (2 * D_INNER) + d];
    }
    ucbf[idx] = f2bf(acc * sigmoid_(acc));
}

// ---------------------------------------------------------------------------
// Selective scan, software-pipelined one row ahead. 32 lanes = 2 channels x
// 16 states; dA/dBu recomputed in-register; output gated y as bf16.
// ---------------------------------------------------------------------------
__global__ __launch_bounds__(256)
void scan_kernel(const __bf16* __restrict__ ucbf, const float* __restrict__ dt,
                 const float* __restrict__ xdbl, const float* __restrict__ xz,
                 const float* __restrict__ A_log, const float* __restrict__ Dp,
                 __bf16* __restrict__ ybf, int rev)
{
    const int tid = threadIdx.x;
    const int n   = tid & 15;
    const int c   = tid >> 4;
    const int d   = blockIdx.x * 16 + c;
    const int b   = blockIdx.y;

    const float Aval = -__expf(A_log[d * D_STATE + n]);
    const float Dd   = Dp[d];
    const size_t rowbase = (size_t)b * SEQ;

    size_t row = rowbase + (rev ? (SEQ - 1) : 0);
    float dtv = dt[row * D_INNER + d];
    float uv  = bf2f(ucbf[row * D_INNER + d]);
    float Bv  = xdbl[row * XDBL_W + DT_RANK + n];
    float Cv  = xdbl[row * XDBL_W + DT_RANK + D_STATE + n];

    float h = 0.0f;
    for (int t = 0; t < SEQ; ++t) {
        const size_t cur = row;
        const float cdt = dtv, cu = uv, cB = Bv, cC = Cv;
        if (t + 1 < SEQ) {                       // prefetch next step
            row = rowbase + (rev ? (SEQ - 2 - t) : (t + 1));
            dtv = dt[row * D_INNER + d];
            uv  = bf2f(ucbf[row * D_INNER + d]);
            Bv  = xdbl[row * XDBL_W + DT_RANK + n];
            Cv  = xdbl[row * XDBL_W + DT_RANK + D_STATE + n];
        }
        h = __expf(cdt * Aval) * h + cdt * cB * cu;
        float yp = h * cC;
        yp += __shfl_xor(yp, 1);
        yp += __shfl_xor(yp, 2);
        yp += __shfl_xor(yp, 4);
        yp += __shfl_xor(yp, 8);
        if (n == 0) {
            const float zv = xz[cur * (2 * D_INNER) + D_INNER + d];
            ybf[cur * D_INNER + d] = f2bf((yp + cu * Dd) * (zv * sigmoid_(zv)));
        }
    }
}

// ---------------------------------------------------------------------------
// LayerNorm over D_MODEL=256: one wave per token row.
// ---------------------------------------------------------------------------
__global__ __launch_bounds__(256)
void layernorm_kernel(const float* __restrict__ h, const float* __restrict__ gamma,
                      const float* __restrict__ beta, float* __restrict__ out)
{
    const int lane = threadIdx.x & 31;
    const int row  = blockIdx.x * 8 + (threadIdx.x >> 5);
    const float* hp = h + (size_t)row * D_MODEL;

    float v[8];
    float s = 0.0f;
#pragma unroll
    for (int i = 0; i < 8; ++i) { v[i] = hp[lane + 32 * i]; s += v[i]; }
#pragma unroll
    for (int m = 16; m >= 1; m >>= 1) s += __shfl_xor(s, m);
    const float mu = s * (1.0f / D_MODEL);

    float var = 0.0f;
#pragma unroll
    for (int i = 0; i < 8; ++i) { const float dv = v[i] - mu; var += dv * dv; }
#pragma unroll
    for (int m = 16; m >= 1; m >>= 1) var += __shfl_xor(var, m);
    const float inv = rsqrtf(var * (1.0f / D_MODEL) + LN_EPS);

    float* op = out + (size_t)row * D_MODEL;
#pragma unroll
    for (int i = 0; i < 8; ++i) {
        const int cidx = lane + 32 * i;
        op[cidx] = (v[i] - mu) * inv * gamma[cidx] + beta[cidx];
    }
}

// ---------------------------------------------------------------------------
extern "C" void kernel_launch(void* const* d_in, const int* in_sizes, int n_in,
                              void* d_out, int out_size, void* d_ws, size_t ws_size,
                              hipStream_t stream)
{
    (void)in_sizes; (void)n_in; (void)out_size; (void)ws_size;

    const float* x        = (const float*)d_in[0];
    const float* f_in_w   = (const float*)d_in[1];
    const float* f_conv_w = (const float*)d_in[2];
    const float* f_conv_b = (const float*)d_in[3];
    const float* f_xprj_w = (const float*)d_in[4];
    const float* f_dt_w   = (const float*)d_in[5];
    const float* f_dt_b   = (const float*)d_in[6];
    const float* f_A_log  = (const float*)d_in[7];
    const float* f_D      = (const float*)d_in[8];
    const float* f_out_w  = (const float*)d_in[9];
    const float* b_in_w   = (const float*)d_in[10];
    const float* b_conv_w = (const float*)d_in[11];
    const float* b_conv_b = (const float*)d_in[12];
    const float* b_xprj_w = (const float*)d_in[13];
    const float* b_dt_w   = (const float*)d_in[14];
    const float* b_dt_b   = (const float*)d_in[15];
    const float* b_A_log  = (const float*)d_in[16];
    const float* b_D      = (const float*)d_in[17];
    const float* b_out_w  = (const float*)d_in[18];
    const float* gamma    = (const float*)d_in[19];
    const float* beta     = (const float*)d_in[20];

    // ---- fp32 workspace region (~224 MB) ----
    float* ws = (float*)d_ws;
    size_t off = 0;
    float* xz_f   = ws + off; off += (size_t)NTOK * 2 * D_INNER;
    float* xz_b   = ws + off; off += (size_t)NTOK * 2 * D_INNER;
    float* dt_f   = ws + off; off += (size_t)NTOK * D_INNER;
    float* dt_b   = ws + off; off += (size_t)NTOK * D_INNER;
    float* xdbl_f = ws + off; off += (size_t)NTOK * XDBL_W;
    float* xdbl_b = ws + off; off += (size_t)NTOK * XDBL_W;
    float* hbuf   = ws + off; off += (size_t)NTOK * D_MODEL;

    // ---- bf16 arena (~81 MB), 16B-aligned (all counts multiples of 8) ----
    __bf16* bfa = (__bf16*)(ws + off);
    size_t bo = 0;
    __bf16* xbf       = bfa + bo; bo += (size_t)NTOK * D_MODEL;
    __bf16* f_in_bf   = bfa + bo; bo += (size_t)2 * D_INNER * D_MODEL;
    __bf16* b_in_bf   = bfa + bo; bo += (size_t)2 * D_INNER * D_MODEL;
    __bf16* f_xprj_bf = bfa + bo; bo += (size_t)XDBL_W * D_INNER;
    __bf16* b_xprj_bf = bfa + bo; bo += (size_t)XDBL_W * D_INNER;
    __bf16* f_dt_bf   = bfa + bo; bo += (size_t)D_INNER * DT_RANK;
    __bf16* b_dt_bf   = bfa + bo; bo += (size_t)D_INNER * DT_RANK;
    __bf16* f_out_bf  = bfa + bo; bo += (size_t)D_MODEL * D_INNER;
    __bf16* b_out_bf  = bfa + bo; bo += (size_t)D_MODEL * D_INNER;
    __bf16* ucbf_f    = bfa + bo; bo += (size_t)NTOK * D_INNER;
    __bf16* ucbf_b    = bfa + bo; bo += (size_t)NTOK * D_INNER;
    __bf16* xdblbf_f  = bfa + bo; bo += (size_t)NTOK * XDBL_W;
    __bf16* xdblbf_b  = bfa + bo; bo += (size_t)NTOK * XDBL_W;
    __bf16* ybf_f     = bfa + bo; bo += (size_t)NTOK * D_INNER;
    __bf16* ybf_b     = bfa + bo; bo += (size_t)NTOK * D_INNER;

    const dim3 blk(256);
    auto cvt = [&](const float* src, __bf16* dst, size_t n) {
        dim3 g((unsigned)((n / 4 + 255) / 256));
        cvt_f32_bf16_kernel<<<g, blk, 0, stream>>>(src, dst, (int)(n / 4));
    };

    // P0: bf16 arena fill
    cvt(x,        xbf,       (size_t)NTOK * D_MODEL);
    cvt(f_in_w,   f_in_bf,   (size_t)2 * D_INNER * D_MODEL);
    cvt(b_in_w,   b_in_bf,   (size_t)2 * D_INNER * D_MODEL);
    cvt(f_xprj_w, f_xprj_bf, (size_t)XDBL_W * D_INNER);
    cvt(b_xprj_w, b_xprj_bf, (size_t)XDBL_W * D_INNER);
    cvt(f_dt_w,   f_dt_bf,   (size_t)D_INNER * DT_RANK);
    cvt(b_dt_w,   b_dt_bf,   (size_t)D_INNER * DT_RANK);
    cvt(f_out_w,  f_out_bf,  (size_t)D_MODEL * D_INNER);
    cvt(b_out_w,  b_out_bf,  (size_t)D_MODEL * D_INNER);

    // K1: in_proj (N=1024, K=256) - TDM-staged FULL GEMM
    {
        dim3 g((2 * D_INNER) / BN, NTOK / BM);
        gemm_bf16_full_kernel<<<g, blk, 0, stream>>>(xbf, f_in_bf, nullptr, nullptr,
                                                     nullptr, xz_f, NTOK, 2 * D_INNER,
                                                     D_MODEL, D_MODEL);
        gemm_bf16_full_kernel<<<g, blk, 0, stream>>>(xbf, b_in_bf, nullptr, nullptr,
                                                     nullptr, xz_b, NTOK, 2 * D_INNER,
                                                     D_MODEL, D_MODEL);
    }

    // K2: causal depthwise conv + SiLU -> bf16 u
    {
        dim3 g((NTOK * D_INNER + 255) / 256);
        conv_silu_kernel<<<g, blk, 0, stream>>>(xz_f, f_conv_w, f_conv_b, ucbf_f, 0);
        conv_silu_kernel<<<g, blk, 0, stream>>>(xz_b, b_conv_w, b_conv_b, ucbf_b, 1);
    }

    // K3: x_proj (N=48, K=512) - SAFE GEMM, fp32 + bf16 outputs
    {
        dim3 g((XDBL_W + BN - 1) / BN, NTOK / BM);
        gemm_bf16_safe_kernel<<<g, blk, 0, stream>>>(ucbf_f, f_xprj_bf, nullptr,
                                                     xdbl_f, xdblbf_f,
                                                     NTOK, XDBL_W, D_INNER, D_INNER, 0);
        gemm_bf16_safe_kernel<<<g, blk, 0, stream>>>(ucbf_b, b_xprj_bf, nullptr,
                                                     xdbl_b, xdblbf_b,
                                                     NTOK, XDBL_W, D_INNER, D_INNER, 0);
    }

    // K4: dt_proj + bias + softplus (N=512, K=16, lda=48) - SAFE GEMM
    {
        dim3 g(D_INNER / BN, NTOK / BM);
        gemm_bf16_safe_kernel<<<g, blk, 0, stream>>>(xdblbf_f, f_dt_bf, f_dt_b,
                                                     dt_f, nullptr,
                                                     NTOK, D_INNER, DT_RANK, XDBL_W, 1);
        gemm_bf16_safe_kernel<<<g, blk, 0, stream>>>(xdblbf_b, b_dt_bf, b_dt_b,
                                                     dt_b, nullptr,
                                                     NTOK, D_INNER, DT_RANK, XDBL_W, 1);
    }

    // K5: selective scans -> bf16 gated y
    {
        dim3 g(D_INNER / 16, BATCH);
        scan_kernel<<<g, blk, 0, stream>>>(ucbf_f, dt_f, xdbl_f, xz_f,
                                           f_A_log, f_D, ybf_f, 0);
        scan_kernel<<<g, blk, 0, stream>>>(ucbf_b, dt_b, xdbl_b, xz_b,
                                           b_A_log, b_D, ybf_b, 1);
    }

    // K6: out_proj dual GEMM + residual -> hbuf
    {
        dim3 g(D_MODEL / BN, NTOK / BM);
        gemm_bf16_full_kernel<<<g, blk, 0, stream>>>(ybf_f, f_out_bf, ybf_b, b_out_bf,
                                                     x, hbuf, NTOK, D_MODEL,
                                                     D_INNER, D_INNER);
    }

    // K7: LayerNorm -> d_out
    {
        dim3 g(NTOK / 8);
        layernorm_kernel<<<g, blk, 0, stream>>>(hbuf, gamma, beta, (float*)d_out);
    }
}
